// customGRU_5866925326690
// MI455X (gfx1250) — matrix operand order repaired and checked by reading, not compiled, not run on hardware
//
#include <hip/hip_runtime.h>
#include <hip/hip_bf16.h>

// S=1024, B=128, V=256, E=128, H=256
#define SEQ 1024
#define BAT 128
#define VOC 256
#define EMB 128
#define HID 256
#define G3  (3*HID)      // 768 packed gate columns (r,z,n)
#define SCAN_BLOCKS 16   // HID/16 column groups, persistent grid

typedef __attribute__((ext_vector_type(16))) __bf16 v16bf;
typedef __attribute__((ext_vector_type(8)))  __bf16 v8bf;
typedef __attribute__((ext_vector_type(8)))  float  v8f;
typedef __attribute__((ext_vector_type(4)))  unsigned v4u;
typedef __attribute__((ext_vector_type(8)))  unsigned v8u;

__device__ __forceinline__ v16bf mk16(v8bf lo, v8bf hi) {
  v16bf r;
#pragma unroll
  for (int i = 0; i < 8; ++i) { r[i] = lo[i]; r[8 + i] = hi[i]; }
  return r;
}

// A fragment (16x32 bf16, MxK): rowp = per-lane row base (row = m0 + (lane&15)).
// element j <-> K = k0 + hi*16 + j
__device__ __forceinline__ v16bf load_a_frag(const __bf16* rowp, int k0, int hi) {
  v8bf c0 = *(const v8bf*)(rowp + k0 + hi * 16);
  v8bf c1 = *(const v8bf*)(rowp + k0 + hi * 16 + 8);
  return mk16(c0, c1);
}

// B fragment (32x16 bf16, KxN) from row-major W[n][k]: element j <-> K = k0 + hi*16 + j
__device__ __forceinline__ v16bf load_b_frag(const __bf16* p) {
  v8bf c0 = *(const v8bf*)p;
  v8bf c1 = *(const v8bf*)(p + 8);
  return mk16(c0, c1);
}

__device__ __forceinline__ v8f wmma_bf16(v16bf a, v16bf b, v8f c) {
  return __builtin_amdgcn_wmma_f32_16x16x32_bf16(false, a, false, b, (short)0, c,
                                                 false, false);
}

// ---------------- fp32 -> bf16 convert ----------------
__global__ void cvt_bf16_kernel(const float* __restrict__ src, __bf16* __restrict__ dst, int n) {
  int i = blockIdx.x * blockDim.x + threadIdx.x;
  if (i < n) dst[i] = (__bf16)src[i];
}

// ---------------- input-side GEMM: gates[r, 0:768] = emb[x[r]] @ Win^T + b ----------------
// Wave computes 16 rows x 64 cols (4 accumulators, 4x A reuse).
__global__ __launch_bounds__(128) void input_gemm_kernel(
    const int* __restrict__ x, const __bf16* __restrict__ emb_bf,
    const __bf16* __restrict__ Win,  // [768][128] bf16 (rx_w; zx_w; nx_w)
    const float* __restrict__ rx_b, const float* __restrict__ zx_b,
    const float* __restrict__ nx_b, float* __restrict__ gates) {
  const int lane = threadIdx.x & 31, wave = threadIdx.x >> 5;
  const int l = lane & 15, hi = lane >> 4;
  const int m0 = blockIdx.x * 16;
  const int n_base = blockIdx.y * 256 + wave * 64;

  const int tok = x[m0 + l];  // per-lane embedding gather
  const __bf16* arow = emb_bf + (size_t)tok * EMB;

  v8f acc[4] = {{}, {}, {}, {}};
#pragma unroll
  for (int k = 0; k < EMB; k += 32) {
    v16bf a = load_a_frag(arow, k, hi);
#pragma unroll
    for (int c = 0; c < 4; ++c) {
      const __bf16* brow = Win + (size_t)(n_base + c * 16 + l) * EMB;
      v16bf b = load_b_frag(brow + k + hi * 16);
      acc[c] = wmma_bf16(a, b, acc[c]);
    }
  }
#pragma unroll
  for (int c = 0; c < 4; ++c) {
    int col = n_base + c * 16 + l;
    float bias = (col < HID) ? rx_b[col]
               : (col < 2 * HID) ? zx_b[col - HID] : nx_b[col - 2 * HID];
#pragma unroll
    for (int i = 0; i < 8; ++i) {
      int row = m0 + hi * 8 + i;
      gates[(size_t)row * G3 + col] = acc[c][i] + bias;
    }
  }
}

// ---- TDM helper: 2D tile (rows x 256 bf16) global -> LDS via tensor_load_to_lds ----
__device__ __forceinline__ void tdm_load_tile_16x256_bf16(unsigned lds_byte_off,
                                                          const __bf16* gsrc) {
  unsigned long long ga = (unsigned long long)(size_t)gsrc;
  v4u g0;
  g0[0] = 1u;                               // count=1, is_restore=0, gather=0
  g0[1] = lds_byte_off;                     // lds_addr
  g0[2] = (unsigned)(ga & 0xffffffffull);   // global_addr[31:0]
  g0[3] = (unsigned)((ga >> 32) & 0x1ffffffull) | 0x80000000u;  // addr[56:32] | type=2
  v8u g1;
  g1[0] = 0x00010000u;        // workgroup_mask=0, data_size=1 (2B)
  g1[1] = 256u << 16;         // tensor_dim0 = 256 (low 16 in [63:48])
  g1[2] = 768u << 16;         // tensor_dim0 hi=0, tensor_dim1[15:0] = 768
  g1[3] = 256u << 16;         // tensor_dim1 hi=0, tile_dim0 = 256
  g1[4] = 16u;                // tile_dim1 = 16, tile_dim2 = 0
  g1[5] = 256u;               // tensor_dim0_stride = 256 (low 32)
  g1[6] = 0u;                 // stride hi, tensor_dim1_stride lo
  g1[7] = 0u;
  asm volatile("tensor_load_to_lds %0, %1" ::"s"(g0), "s"(g1) : "memory");
}

// ---------------- persistent GRU scan ----------------
__global__ __launch_bounds__(256) void gru_scan_kernel(
    const __bf16* __restrict__ h0_bf, __bf16* __restrict__ hs,
    const float* __restrict__ gates,
    const __bf16* __restrict__ Wh,  // [768][256] bf16 (rh_w; zh_w; nh_w)
    const float* __restrict__ rh_b, const float* __restrict__ zh_b,
    const float* __restrict__ nh_b, const float* __restrict__ hprev,
    float* __restrict__ hfinal, unsigned* __restrict__ bar) {
  __shared__ __attribute__((aligned(16))) __bf16 lwh[3][16][HID];  // 24 KB weight cols

  const int tid = threadIdx.x, lane = tid & 31, wave = tid >> 5;
  const int l = lane & 15, hi = lane >> 4;
  const int j0 = blockIdx.x * 16;  // hidden-column group

  // Stage Wh columns j0..j0+15 (3 gate blocks) into LDS with the Tensor Data Mover.
  if (wave == 0) {
    unsigned lds_base = (unsigned)(size_t)(&lwh[0][0][0]);  // low 32 bits = LDS offset
#pragma unroll
    for (int g = 0; g < 3; ++g)
      tdm_load_tile_16x256_bf16(lds_base + (unsigned)g * 16 * HID * 2,
                                Wh + (size_t)(g * HID + j0) * HID);
    __builtin_amdgcn_s_wait_tensorcnt(0);
  }
  __syncthreads();

  const int m0 = wave * 16;  // batch-row tile (8 waves cover B=128)
  const int col = j0 + l;
  const float br = rh_b[col], bz = zh_b[col], bn = nh_b[col];

  v8f h;  // register-resident h: each (b, j) owned by one thread for all steps
#pragma unroll
  for (int i = 0; i < 8; ++i) h[i] = hprev[(size_t)(m0 + hi * 8 + i) * HID + col];

  const __bf16* hp = h0_bf;
  for (int s = 0; s < SEQ; ++s) {
    const __bf16* arow = hp + (size_t)(m0 + l) * HID;
    if (s + 1 < SEQ) {  // warm next step's gate rows while WMMAs run
      const float* pre = gates + (size_t)(s + 1) * BAT * G3 + (size_t)(m0 + hi * 8) * G3 + col;
      __builtin_prefetch(pre, 0, 0);
    }
    v8f ar = {}, az = {}, an = {};
#pragma unroll
    for (int k = 0; k < 8; ++k) {
      v16bf a  = load_a_frag(arow, k * 32, hi);
      v16bf b0 = load_b_frag(&lwh[0][l][k * 32 + hi * 16]);
      v16bf b1 = load_b_frag(&lwh[1][l][k * 32 + hi * 16]);
      v16bf b2 = load_b_frag(&lwh[2][l][k * 32 + hi * 16]);
      ar = wmma_bf16(a, b0, ar);
      az = wmma_bf16(a, b1, az);
      an = wmma_bf16(a, b2, an);
    }
    // fc_t = log(ct/(1-ct)), ct = 1 - (c/(c+1))^2, c = s+1
    float c = (float)(s + 1);
    float q = c / (c + 1.0f);
    float ctv = 1.0f - q * q;
    float fc = __logf(ctv / (1.0f - ctv));

    const float* grow = gates + (size_t)s * BAT * G3;
#pragma unroll
    for (int i = 0; i < 8; ++i) {
      int row = m0 + hi * 8 + i;
      const float* gp = grow + (size_t)row * G3;
      float rx = gp[col], zx = gp[HID + col], nx = gp[2 * HID + col];
      float rt = 1.0f / (1.0f + __expf(-(rx + ar[i] + br)));
      float zt = 1.0f / (1.0f + __expf(-(zx + az[i] + bz + fc)));
      float nt = tanhf(nx + rt * (an[i] + bn));
      float hn = (1.0f - zt) * h[i] + zt * nt;
      h[i] = hn;
      hs[(size_t)s * BAT * HID + (size_t)row * HID + col] = (__bf16)hn;
    }

    // device-wide step barrier (monotonic counter, agent scope)
    __threadfence();
    __syncthreads();
    if (tid == 0) {
      __hip_atomic_fetch_add(bar, 1u, __ATOMIC_ACQ_REL, __HIP_MEMORY_SCOPE_AGENT);
      unsigned target = (unsigned)(s + 1) * SCAN_BLOCKS;
      while (__hip_atomic_load(bar, __ATOMIC_ACQUIRE, __HIP_MEMORY_SCOPE_AGENT) < target)
        __builtin_amdgcn_s_sleep(2);
    }
    __syncthreads();
    hp = hs + (size_t)s * BAT * HID;
  }
#pragma unroll
  for (int i = 0; i < 8; ++i)
    hfinal[(size_t)(m0 + hi * 8 + i) * HID + col] = h[i];
}

// ---------------- output dense GEMM: logits = hs @ dense_w^T + b ----------------
// Wave computes 16 rows x 64 cols (4 accumulators); block of 4 waves covers all 256 cols.
__global__ __launch_bounds__(128) void dense_gemm_kernel(
    const __bf16* __restrict__ hs, const __bf16* __restrict__ Wd,  // [256][256] bf16
    const float* __restrict__ db, float* __restrict__ out) {
  const int lane = threadIdx.x & 31, wave = threadIdx.x >> 5;
  const int l = lane & 15, hi = lane >> 4;
  const int m0 = blockIdx.x * 16;
  const int n_base = wave * 64;

  const __bf16* arow = hs + (size_t)(m0 + l) * HID;
  v8f acc[4] = {{}, {}, {}, {}};
#pragma unroll
  for (int k = 0; k < HID; k += 32) {
    v16bf a = load_a_frag(arow, k, hi);
#pragma unroll
    for (int c = 0; c < 4; ++c) {
      const __bf16* brow = Wd + (size_t)(n_base + c * 16 + l) * HID;
      v16bf b = load_b_frag(brow + k + hi * 16);
      acc[c] = wmma_bf16(a, b, acc[c]);
    }
  }
#pragma unroll
  for (int c = 0; c < 4; ++c) {
    int col = n_base + c * 16 + l;
    float bias = db[col];
#pragma unroll
    for (int i = 0; i < 8; ++i) {
      int row = m0 + hi * 8 + i;
      out[(size_t)row * VOC + col] = acc[c][i] + bias;
    }
  }
}

extern "C" void kernel_launch(void* const* d_in, const int* in_sizes, int n_in,
                              void* d_out, int out_size, void* d_ws, size_t ws_size,
                              hipStream_t stream) {
  const int*   x      = (const int*)d_in[0];
  const float* hprev  = (const float*)d_in[1];
  const float* emb    = (const float*)d_in[2];
  const float* rx_w   = (const float*)d_in[3];
  const float* rx_b   = (const float*)d_in[4];
  const float* rh_w   = (const float*)d_in[5];
  const float* rh_b   = (const float*)d_in[6];
  const float* zx_w   = (const float*)d_in[7];
  const float* zx_b   = (const float*)d_in[8];
  const float* zh_w   = (const float*)d_in[9];
  const float* zh_b   = (const float*)d_in[10];
  const float* nx_w   = (const float*)d_in[11];
  const float* nx_b   = (const float*)d_in[12];
  const float* nh_w   = (const float*)d_in[13];
  const float* nh_b   = (const float*)d_in[14];
  const float* dw     = (const float*)d_in[15];
  const float* db     = (const float*)d_in[16];
  float* out = (float*)d_out;

  // workspace carve-up (256B aligned)
  char* ws = (char*)d_ws;
  size_t off = 0;
  auto take = [&](size_t bytes) { void* p = ws + off; off = (off + bytes + 255) & ~(size_t)255; return p; };
  __bf16* emb_bf  = (__bf16*)take((size_t)VOC * EMB * 2);
  __bf16* Win_bf  = (__bf16*)take((size_t)G3 * EMB * 2);
  __bf16* Wh_bf   = (__bf16*)take((size_t)G3 * HID * 2);
  __bf16* Wd_bf   = (__bf16*)take((size_t)VOC * HID * 2);
  __bf16* h0_bf   = (__bf16*)take((size_t)BAT * HID * 2);
  unsigned* bar   = (unsigned*)take(256);
  float*  gates   = (float*)take((size_t)SEQ * BAT * G3 * 4);   // 402 MB
  __bf16* hs      = (__bf16*)take((size_t)SEQ * BAT * HID * 2); // 67 MB

  hipMemsetAsync(bar, 0, 256, stream);

  // bf16 weight/embedding packing
  auto cvt = [&](const float* s, __bf16* d, int n) {
    cvt_bf16_kernel<<<(n + 255) / 256, 256, 0, stream>>>(s, d, n);
  };
  cvt(emb,  emb_bf,              VOC * EMB);
  cvt(rx_w, Win_bf + 0 * HID * EMB, HID * EMB);
  cvt(zx_w, Win_bf + 1 * HID * EMB, HID * EMB);
  cvt(nx_w, Win_bf + 2 * HID * EMB, HID * EMB);
  cvt(rh_w, Wh_bf + 0 * HID * HID, HID * HID);
  cvt(zh_w, Wh_bf + 1 * HID * HID, HID * HID);
  cvt(nh_w, Wh_bf + 2 * HID * HID, HID * HID);
  cvt(dw,   Wd_bf,               VOC * HID);
  cvt(hprev, h0_bf,              BAT * HID);

  // fused embedding + 3-gate input GEMM  (grid: 8192 row tiles x 3 column blocks)
  input_gemm_kernel<<<dim3(SEQ * BAT / 16, G3 / 256), 128, 0, stream>>>(
      x, emb_bf, Win_bf, rx_b, zx_b, nx_b, gates);

  // persistent recurrence (h_final written directly to tail of d_out)
  float* hfinal = out + (size_t)SEQ * BAT * VOC;
  gru_scan_kernel<<<SCAN_BLOCKS, 256, 0, stream>>>(
      h0_bf, hs, gates, Wh_bf, rh_b, zh_b, nh_b, hprev, hfinal, bar);

  // output projection (block of 4 waves covers all 256 output cols)
  dense_gemm_kernel<<<SEQ * BAT / 16, 128, 0, stream>>>(hs, Wd_bf, db, out);
}
